// TopKPooling_59811714564729
// MI455X (gfx1250) — matrix operand order repaired and checked by reading, not compiled, not run on hardware
//
#include <hip/hip_runtime.h>
#include <hip/hip_bf16.h>

// Problem constants (from the reference): N=16384 nodes, M=4096 edges, D=256.
#define N_NODES 16384
#define M_EDGES 4096
#define D_FEAT  256
#define K_TOP   8192          // ceil(0.5 * N)
#define NBINS   1024

typedef __attribute__((ext_vector_type(16))) __bf16 v16bf;
typedef __attribute__((ext_vector_type(8)))  float  v8f;
typedef __attribute__((ext_vector_type(4)))  unsigned int ui32x4;
typedef __attribute__((ext_vector_type(8)))  int    i32x8;
typedef __attribute__((ext_vector_type(4)))  int    i32x4;

union FragBF { uint4 u[2]; v16bf v; };

__device__ __forceinline__ __bf16 to_bf(float f) { return (__bf16)f; }

// ---------------------------------------------------------------------------
// zero helper
// ---------------------------------------------------------------------------
__global__ void zero_kernel(float* __restrict__ p, int n) {
  int i = blockIdx.x * blockDim.x + threadIdx.x;
  if (i < n) p[i] = 0.0f;
}

// ---------------------------------------------------------------------------
// W (256x256 fp32) -> bf16 once; it is L2-resident for GEMM2b.
// ---------------------------------------------------------------------------
__global__ void wconv_kernel(const float* __restrict__ Wm, __bf16* __restrict__ wbf) {
  int i = blockIdx.x * 256 + threadIdx.x;
  wbf[i] = to_bf(Wm[i]);
}

// ---------------------------------------------------------------------------
// Fused degree kernel: one pass over incidence computes
//   deg_v[n] = max(sum_e inc[n][e], 1)   (exclusive rows per block, direct store)
//   deg_e_acc[e] += partial column sums  (global fp32 atomics)
// 128 rows per block, 256 threads; thread t owns columns t, t+256, ... (16 cols).
// ---------------------------------------------------------------------------
__global__ __launch_bounds__(256) void degrees_kernel(const float* __restrict__ inc,
                                                      float* __restrict__ deg_e_acc,
                                                      float* __restrict__ deg_v) {
  __shared__ float rowsum[128];
  const int tid = threadIdx.x, lane = tid & 31;
  const int n0 = blockIdx.x * 128;
  float colacc[16];
#pragma unroll
  for (int j = 0; j < 16; ++j) colacc[j] = 0.0f;
  if (tid < 128) rowsum[tid] = 0.0f;
  __syncthreads();
  for (int r = 0; r < 128; ++r) {
    const float* row = &inc[(size_t)(n0 + r) * M_EDGES];
    float part = 0.0f;
#pragma unroll
    for (int j = 0; j < 16; ++j) {
      float v = row[tid + j * 256];   // coalesced across the block
      part += v;
      colacc[j] += v;
    }
#pragma unroll
    for (int off = 16; off; off >>= 1) part += __shfl_xor(part, off);
    if (lane == 0) atomicAdd(&rowsum[r], part);
  }
  __syncthreads();
  if (tid < 128) deg_v[n0 + tid] = fmaxf(rowsum[tid], 1.0f);
#pragma unroll
  for (int j = 0; j < 16; ++j) atomicAdd(&deg_e_acc[tid + j * 256], colacc[j]);
}

// ---------------------------------------------------------------------------
// GEMM1: x_acc[M_EDGES][D] += inc^T @ feat  (split-K, fp32 atomic accumulate)
//   A[m][k] = inc[k*M + m]   (m-contiguous in memory -> coalesced loads)
//   B[k][n] = feat[k*D + n]
// bf16 WMMA 16x16x32, WG tile 128x128, 8 waves in a 4x2 grid (wave tile 32x64).
// ---------------------------------------------------------------------------
__global__ __launch_bounds__(256) void gemm1_kernel(const float* __restrict__ inc,
                                                    const float* __restrict__ feat,
                                                    float* __restrict__ x_acc) {
  __shared__ __bf16 As[128][40];   // [m][k], row stride 80B (16B aligned)
  __shared__ __bf16 Bs[128][40];   // [n][k]
  const int tid = threadIdx.x;
  const int lane = tid & 31;
  const int wave = tid >> 5;
  const int warpM = wave & 3;
  const int warpN = wave >> 2;
  const int m0 = blockIdx.x * 128;              // edge block
  const int n0 = blockIdx.y * 128;              // feature block
  const int kbase = blockIdx.z * (N_NODES / 8); // split-K chunk

  v8f c[2][4] = {};

  for (int ks = 0; ks < (N_NODES / 8) / 32; ++ks) {
    const int k0 = kbase + ks * 32;
    __builtin_prefetch(&inc[(size_t)(k0 + 32) * M_EDGES + m0], 0, 0);
    __builtin_prefetch(&feat[(size_t)(k0 + 32) * D_FEAT + n0], 0, 0);
    // A tile 32(k) x 128(m), m-fast float4 loads, transposed bf16 LDS store
#pragma unroll
    for (int i = 0; i < 4; ++i) {
      int t = tid + i * 256;
      int kk = t >> 5;
      int mm = (t & 31) << 2;
      float4 v = *reinterpret_cast<const float4*>(&inc[(size_t)(k0 + kk) * M_EDGES + m0 + mm]);
      As[mm + 0][kk] = to_bf(v.x); As[mm + 1][kk] = to_bf(v.y);
      As[mm + 2][kk] = to_bf(v.z); As[mm + 3][kk] = to_bf(v.w);
    }
    // B tile 32(k) x 128(n), n-fast float4 loads, transposed bf16 LDS store
#pragma unroll
    for (int i = 0; i < 4; ++i) {
      int t = tid + i * 256;
      int kk = t >> 5;
      int nn = (t & 31) << 2;
      float4 v = *reinterpret_cast<const float4*>(&feat[(size_t)(k0 + kk) * D_FEAT + n0 + nn]);
      Bs[nn + 0][kk] = to_bf(v.x); Bs[nn + 1][kk] = to_bf(v.y);
      Bs[nn + 2][kk] = to_bf(v.z); Bs[nn + 3][kk] = to_bf(v.w);
    }
    __syncthreads();
    const int khalf = lane >> 4;
    const int l16 = lane & 15;
    FragBF a[2];
#pragma unroll
    for (int im = 0; im < 2; ++im) {
      int m = warpM * 32 + im * 16 + l16;
      a[im].u[0] = *reinterpret_cast<const uint4*>(&As[m][khalf * 8]);
      a[im].u[1] = *reinterpret_cast<const uint4*>(&As[m][16 + khalf * 8]);
    }
#pragma unroll
    for (int in = 0; in < 4; ++in) {
      int n = warpN * 64 + in * 16 + l16;
      FragBF b;
      b.u[0] = *reinterpret_cast<const uint4*>(&Bs[n][khalf * 8]);
      b.u[1] = *reinterpret_cast<const uint4*>(&Bs[n][16 + khalf * 8]);
#pragma unroll
      for (int im = 0; im < 2; ++im)
        c[im][in] = __builtin_amdgcn_wmma_f32_16x16x32_bf16(
            false, a[im].v, false, b.v, (short)0, c[im][in], false, false);
    }
    __syncthreads();
  }
  // split-K epilogue: atomic accumulate into fp32 buffer
#pragma unroll
  for (int im = 0; im < 2; ++im)
#pragma unroll
    for (int in = 0; in < 4; ++in) {
      int col = n0 + warpN * 64 + in * 16 + (lane & 15);
#pragma unroll
      for (int r = 0; r < 8; ++r) {
        int row = m0 + warpM * 32 + im * 16 + r + ((lane >> 4) << 3);
        atomicAdd(&x_acc[(size_t)row * D_FEAT + col], c[im][in][r]);
      }
    }
}

// ---------------------------------------------------------------------------
// x_edge_bf16 = x_acc / max(deg_e, 1)
// ---------------------------------------------------------------------------
__global__ void finalize_x_kernel(const float* __restrict__ x_acc,
                                  const float* __restrict__ deg_e_acc,
                                  __bf16* __restrict__ xe) {
  int i = blockIdx.x * 256 + threadIdx.x;
  int e = i >> 8;
  xe[i] = to_bf(x_acc[i] / fmaxf(deg_e_acc[e], 1.0f));
}

// ---------------------------------------------------------------------------
// GEMM2a: agg[N][D] = (inc @ x_edge) / deg_v, bf16 output
//   A[m][k] = inc[m*M + k]  (k-contiguous)
//   B[k][n] = xe[k*D + n]   (already bf16)
// ---------------------------------------------------------------------------
__global__ __launch_bounds__(256) void gemm2a_kernel(const float* __restrict__ inc,
                                                     const __bf16* __restrict__ xe,
                                                     const float* __restrict__ deg_v,
                                                     __bf16* __restrict__ agg) {
  __shared__ __bf16 As[128][40];
  __shared__ __bf16 Bs[128][40];
  const int tid = threadIdx.x;
  const int lane = tid & 31;
  const int wave = tid >> 5;
  const int warpM = wave & 3;
  const int warpN = wave >> 2;
  const int m0 = blockIdx.x * 128;  // node block
  const int n0 = blockIdx.y * 128;  // feature block

  v8f c[2][4] = {};

  for (int ks = 0; ks < M_EDGES / 32; ++ks) {
    const int k0 = ks * 32;
    __builtin_prefetch(&inc[(size_t)m0 * M_EDGES + k0 + 32], 0, 0);
    // A tile 128(m) x 32(k), k-fast float4 loads, natural-layout bf16 store
#pragma unroll
    for (int i = 0; i < 4; ++i) {
      int t = tid + i * 256;
      int m = t >> 3;
      int k4 = (t & 7) << 2;
      float4 v = *reinterpret_cast<const float4*>(&inc[(size_t)(m0 + m) * M_EDGES + k0 + k4]);
      As[m][k4 + 0] = to_bf(v.x); As[m][k4 + 1] = to_bf(v.y);
      As[m][k4 + 2] = to_bf(v.z); As[m][k4 + 3] = to_bf(v.w);
    }
    // B tile 32(k) x 128(n) bf16, transposed LDS store
#pragma unroll
    for (int i = 0; i < 2; ++i) {
      int t = tid + i * 256;
      int kk = t >> 4;
      int n8 = (t & 15) << 3;
      uint4 v = *reinterpret_cast<const uint4*>(&xe[(size_t)(k0 + kk) * D_FEAT + n0 + n8]);
      const __bf16* pv = reinterpret_cast<const __bf16*>(&v);
#pragma unroll
      for (int j = 0; j < 8; ++j) Bs[n8 + j][kk] = pv[j];
    }
    __syncthreads();
    const int khalf = lane >> 4;
    const int l16 = lane & 15;
    FragBF a[2];
#pragma unroll
    for (int im = 0; im < 2; ++im) {
      int m = warpM * 32 + im * 16 + l16;
      a[im].u[0] = *reinterpret_cast<const uint4*>(&As[m][khalf * 8]);
      a[im].u[1] = *reinterpret_cast<const uint4*>(&As[m][16 + khalf * 8]);
    }
#pragma unroll
    for (int in = 0; in < 4; ++in) {
      int n = warpN * 64 + in * 16 + l16;
      FragBF b;
      b.u[0] = *reinterpret_cast<const uint4*>(&Bs[n][khalf * 8]);
      b.u[1] = *reinterpret_cast<const uint4*>(&Bs[n][16 + khalf * 8]);
#pragma unroll
      for (int im = 0; im < 2; ++im)
        c[im][in] = __builtin_amdgcn_wmma_f32_16x16x32_bf16(
            false, a[im].v, false, b.v, (short)0, c[im][in], false, false);
    }
    __syncthreads();
  }
#pragma unroll
  for (int im = 0; im < 2; ++im)
#pragma unroll
    for (int r = 0; r < 8; ++r) {
      int row = m0 + warpM * 32 + im * 16 + r + ((lane >> 4) << 3);
      float inv = 1.0f / deg_v[row];   // deg_v pre-clamped to >= 1
#pragma unroll
      for (int in = 0; in < 4; ++in) {
        int col = n0 + warpN * 64 + in * 16 + (lane & 15);
        agg[(size_t)row * D_FEAT + col] = to_bf(c[im][in][r] * inv);
      }
    }
}

// ---------------------------------------------------------------------------
// GEMM2b: emb[N][D] = agg @ W + b   (fp32 out; K = 256, 8 k-steps)
// A-tile (bf16, no transpose needed) is fetched by the Tensor Data Mover:
// one wave issues TENSOR_LOAD_TO_LDS with a 2D D# whose pad fields reproduce
// the 80B padded LDS row stride (64B data + 16B pad), then s_wait_tensorcnt
// + workgroup barrier publishes the tile.
// ---------------------------------------------------------------------------
__global__ __launch_bounds__(256) void gemm2b_kernel(const __bf16* __restrict__ agg,
                                                     const __bf16* __restrict__ wbf,
                                                     const float* __restrict__ bias,
                                                     float* __restrict__ emb) {
  __shared__ __bf16 As[128][40];
  __shared__ __bf16 Bs[128][40];
  const int tid = threadIdx.x;
  const int lane = tid & 31;
  const int wave = tid >> 5;
  const int warpM = wave & 3;
  const int warpN = wave >> 2;
  const int m0 = blockIdx.x * 128;
  const int n0 = blockIdx.y * 128;
  const bool tdm_wave = (wave == 0);
  const unsigned lds_base = (unsigned)(uintptr_t)&As[0][0];

  v8f c[2][4] = {};

  for (int ks = 0; ks < D_FEAT / 32; ++ks) {
    const int k0 = ks * 32;
    // ---- TDM: DMA the 128x32 bf16 A tile into LDS with padded row stride.
    if (tdm_wave) {
      unsigned long long ga =
          (unsigned long long)(uintptr_t)(agg + (size_t)m0 * D_FEAT + k0);
      ui32x4 g0;
      g0[0] = 1u;                                   // count = 1 valid descriptor
      g0[1] = lds_base;                             // lds_addr
      g0[2] = (unsigned)ga;                         // global_addr[31:0]
      g0[3] = (unsigned)(ga >> 32) | (2u << 30);    // global_addr[56:32] | type=2
      i32x8 g1;
      g1[0] = (int)((1u << 16)      // data_size = 2 bytes
                  | (1u << 20)      // pad_enable
                  | (3u << 22)      // pad_interval: every 16 DWORDs (64B)
                  | (3u << 25));    // pad_amount: 4 DWORDs (16B) -> 80B stride
      g1[1] = (int)(256u << 16);    // tensor_dim0 = 256 (bits [79:48], low part)
      g1[2] = (int)(0x4000u << 16); // tensor_dim1 = 16384 (bits [111:80], low part)
      g1[3] = (int)(32u << 16);     // tile_dim0 = 32 (bits [127:112])
      g1[4] = 128;                  // tile_dim1 = 128 (bits [143:128])
      g1[5] = 256;                  // tensor_dim0_stride = 256 (bits [207:160])
      g1[6] = 0;
      g1[7] = 0;
      i32x4 zz4 = {0, 0, 0, 0};               // groups 2/3 unused (2D tensor)
      i32x8 zz8 = {0, 0, 0, 0, 0, 0, 0, 0};   // extra group (6-arg toolchain form)
      __builtin_amdgcn_tensor_load_to_lds(g0, g1, zz4, zz4, zz8, 0);
    }
    // ---- B tile 32(k) x 128(n) bf16, transposed store (manual path)
#pragma unroll
    for (int i = 0; i < 2; ++i) {
      int t = tid + i * 256;
      int kk = t >> 4;
      int n8 = (t & 15) << 3;
      uint4 v = *reinterpret_cast<const uint4*>(&wbf[(size_t)(k0 + kk) * D_FEAT + n0 + n8]);
      const __bf16* pv = reinterpret_cast<const __bf16*>(&v);
#pragma unroll
      for (int j = 0; j < 8; ++j) Bs[n8 + j][kk] = pv[j];
    }
    if (tdm_wave) __builtin_amdgcn_s_wait_tensorcnt(0);
    __syncthreads();
    const int khalf = lane >> 4;
    const int l16 = lane & 15;
    FragBF a[2];
#pragma unroll
    for (int im = 0; im < 2; ++im) {
      int m = warpM * 32 + im * 16 + l16;
      a[im].u[0] = *reinterpret_cast<const uint4*>(&As[m][khalf * 8]);
      a[im].u[1] = *reinterpret_cast<const uint4*>(&As[m][16 + khalf * 8]);
    }
#pragma unroll
    for (int in = 0; in < 4; ++in) {
      int n = warpN * 64 + in * 16 + l16;
      FragBF b;
      b.u[0] = *reinterpret_cast<const uint4*>(&Bs[n][khalf * 8]);
      b.u[1] = *reinterpret_cast<const uint4*>(&Bs[n][16 + khalf * 8]);
#pragma unroll
      for (int im = 0; im < 2; ++im)
        c[im][in] = __builtin_amdgcn_wmma_f32_16x16x32_bf16(
            false, a[im].v, false, b.v, (short)0, c[im][in], false, false);
    }
    __syncthreads();
  }
#pragma unroll
  for (int im = 0; im < 2; ++im)
#pragma unroll
    for (int in = 0; in < 4; ++in) {
      int col = n0 + warpN * 64 + in * 16 + (lane & 15);
      float bv = bias[col];
#pragma unroll
      for (int r = 0; r < 8; ++r) {
        int row = m0 + warpM * 32 + im * 16 + r + ((lane >> 4) << 3);
        emb[(size_t)row * D_FEAT + col] = c[im][in][r] + bv;
      }
    }
}

// ---------------------------------------------------------------------------
// scores[n] = sigmoid(emb[n] . proj), masked to -inf; histogram for selection.
// One wave32 per node.
// ---------------------------------------------------------------------------
__global__ __launch_bounds__(256) void score_kernel(const float* __restrict__ emb,
                                                    const float* __restrict__ proj,
                                                    const unsigned char* __restrict__ node_mask,
                                                    float* __restrict__ scores,
                                                    int* __restrict__ hist) {
  const int wave = threadIdx.x >> 5, lane = threadIdx.x & 31;
  const int node = blockIdx.x * 8 + wave;
  float acc = 0.0f;
#pragma unroll
  for (int j = 0; j < 8; ++j) {
    int d = lane + j * 32;
    acc += emb[(size_t)node * D_FEAT + d] * proj[d];
  }
#pragma unroll
  for (int off = 16; off; off >>= 1) acc += __shfl_xor(acc, off);
  if (lane == 0) {
    float s = 1.0f / (1.0f + __expf(-acc));
    if (!node_mask[node]) s = -INFINITY;
    scores[node] = s;
    if (s > -1e30f) {
      int b = (int)(s * (float)NBINS);
      b = min(max(b, 0), NBINS - 1);
      atomicAdd(&hist[b], 1);
    }
  }
}

// ---------------------------------------------------------------------------
// Find threshold bin T and in-bin quota for exact top-K selection.
// ---------------------------------------------------------------------------
__global__ void threshold_kernel(const int* __restrict__ hist, int* __restrict__ counters) {
  int total = 0, T = 0, quota = K_TOP;
  for (int b = NBINS - 1; b >= 0; --b) {
    int h = hist[b];
    if (total + h >= K_TOP) { T = b; quota = K_TOP - total; break; }
    total += h;
  }
  counters[0] = T;
  counters[1] = quota;   // atomic ticket counter for ties in bin T
}

// ---------------------------------------------------------------------------
// Per-node: decide membership (ticketed ties), write node-mask output,
// write gated_features = feat * score. One block (256 threads) per node.
// ---------------------------------------------------------------------------
__global__ __launch_bounds__(256) void gate_kernel(const float* __restrict__ feat,
                                                   const float* __restrict__ scores,
                                                   int* __restrict__ counters,
                                                   unsigned char* __restrict__ new_mask,
                                                   float* __restrict__ out_gated,
                                                   float* __restrict__ out_nodemask) {
  const int node = blockIdx.x;
  __shared__ float ss;
  if (threadIdx.x == 0) {
    float s = scores[node];
    int sel = 0;
    if (s > -1e30f) {
      int b = (int)(s * (float)NBINS);
      b = min(max(b, 0), NBINS - 1);
      int T = counters[0];
      if (b > T) sel = 1;
      else if (b == T) { int t = atomicSub(&counters[1], 1); sel = (t > 0) ? 1 : 0; }
    }
    ss = s;
    new_mask[node] = (unsigned char)sel;
    out_nodemask[node] = sel ? 1.0f : 0.0f;
  }
  __syncthreads();
  size_t idx = (size_t)node * D_FEAT + threadIdx.x;
  out_gated[idx] = feat[idx] * ss;
}

// ---------------------------------------------------------------------------
// new_edge_mask[e] = (any active node touches edge e) && edge_mask[e].
// One wave per 32 edges; ballot early-exit (expected ~10% of full pass).
// ---------------------------------------------------------------------------
__global__ __launch_bounds__(256) void edge_kernel(const float* __restrict__ inc,
                                                   const unsigned char* __restrict__ new_mask,
                                                   const unsigned char* __restrict__ edge_mask,
                                                   float* __restrict__ out_edge) {
  const int wave = threadIdx.x >> 5, lane = threadIdx.x & 31;
  const int e = (blockIdx.x * 8 + wave) * 32 + lane;
  bool found = false;
  for (int n = 0; n < N_NODES; ++n) {
    if (new_mask[n]) {
      found = found || (inc[(size_t)n * M_EDGES + e] != 0.0f);
      if (__all(found ? 1 : 0)) break;
    }
  }
  out_edge[e] = (found && edge_mask[e]) ? 1.0f : 0.0f;
}

// ---------------------------------------------------------------------------
extern "C" void kernel_launch(void* const* d_in, const int* in_sizes, int n_in,
                              void* d_out, int out_size, void* d_ws, size_t ws_size,
                              hipStream_t stream) {
  (void)in_sizes; (void)n_in; (void)out_size; (void)ws_size;
  const float* feat          = (const float*)d_in[0];
  const float* inc           = (const float*)d_in[1];
  const unsigned char* nmask = (const unsigned char*)d_in[2];
  const unsigned char* emask = (const unsigned char*)d_in[3];
  const float* Wm            = (const float*)d_in[4];
  const float* bias          = (const float*)d_in[5];
  const float* proj          = (const float*)d_in[6];

  char* base = (char*)d_ws;
  size_t off = 0;
  auto carve = [&](size_t bytes) -> void* {
    void* p = base + off;
    off = (off + bytes + 255) & ~(size_t)255;
    return p;
  };
  float*  deg_e_acc = (float*)carve((size_t)M_EDGES * 4);
  float*  deg_v     = (float*)carve((size_t)N_NODES * 4);
  float*  x_acc     = (float*)carve((size_t)M_EDGES * D_FEAT * 4);
  __bf16* xe        = (__bf16*)carve((size_t)M_EDGES * D_FEAT * 2);
  __bf16* agg       = (__bf16*)carve((size_t)N_NODES * D_FEAT * 2);
  __bf16* wbf       = (__bf16*)carve((size_t)D_FEAT * D_FEAT * 2);
  float*  emb       = (float*)carve((size_t)N_NODES * D_FEAT * 4);
  float*  scores    = (float*)carve((size_t)N_NODES * 4);
  int*    hist      = (int*)carve((size_t)NBINS * 4);
  int*    counters  = (int*)carve(64);
  unsigned char* new_mask = (unsigned char*)carve((size_t)N_NODES);

  float* out_gated    = (float*)d_out;
  float* out_nodemask = out_gated + (size_t)N_NODES * D_FEAT;
  float* out_edgemask = out_nodemask + N_NODES;

  // 1) zero accumulators (must re-zero every call: graph replays don't reset ws)
  zero_kernel<<<(M_EDGES * D_FEAT + 255) / 256, 256, 0, stream>>>(x_acc, M_EDGES * D_FEAT);
  zero_kernel<<<(M_EDGES + 255) / 256, 256, 0, stream>>>(deg_e_acc, M_EDGES);
  zero_kernel<<<(NBINS + 255) / 256, 256, 0, stream>>>((float*)hist, NBINS);

  // 2) W -> bf16
  wconv_kernel<<<D_FEAT * D_FEAT / 256, 256, 0, stream>>>(Wm, wbf);

  // 3) degrees (one fused pass over incidence)
  degrees_kernel<<<N_NODES / 128, 256, 0, stream>>>(inc, deg_e_acc, deg_v);

  // 4) GEMM1: inc^T @ feat (split-K = 8 -> 512 workgroups)
  gemm1_kernel<<<dim3(M_EDGES / 128, D_FEAT / 128, 8), 256, 0, stream>>>(inc, feat, x_acc);

  // 5) x_edge = x_acc / deg_e (bf16)
  finalize_x_kernel<<<M_EDGES * D_FEAT / 256, 256, 0, stream>>>(x_acc, deg_e_acc, xe);

  // 6) GEMM2a: agg = (inc @ x_edge) / deg_v (bf16)
  gemm2a_kernel<<<dim3(N_NODES / 128, D_FEAT / 128), 256, 0, stream>>>(inc, xe, deg_v, agg);

  // 7) GEMM2b: emb = agg @ W + b (fp32), A-tile via Tensor Data Mover
  gemm2b_kernel<<<dim3(N_NODES / 128, D_FEAT / 128), 256, 0, stream>>>(agg, wbf, bias, emb);

  // 8) scores + histogram
  score_kernel<<<N_NODES / 8, 256, 0, stream>>>(emb, proj, nmask, scores, hist);

  // 9) top-K threshold
  threshold_kernel<<<1, 1, 0, stream>>>(hist, counters);

  // 10) gating + node mask
  gate_kernel<<<N_NODES, 256, 0, stream>>>(feat, scores, counters, new_mask,
                                           out_gated, out_nodemask);

  // 11) edge mask (early-exit scan over incidence columns)
  edge_kernel<<<M_EDGES / 256, 256, 0, stream>>>(inc, new_mask, emask, out_edgemask);
}
